// SPINN_67817533604434
// MI455X (gfx1250) — matrix operand order repaired and checked by reading, not compiled, not run on hardware
//
#include <hip/hip_runtime.h>

typedef __attribute__((ext_vector_type(16))) __bf16 v16bf;
typedef __attribute__((ext_vector_type(8)))  __bf16 v8bf;
typedef __attribute__((ext_vector_type(8)))  float  v8f;

// Problem constants
#define BN 128
#define LN 64
#define EN 512
#define HN 512
#define TN 127   // 2*L-1

// ---- workspace layout (bytes) ----
static constexpr size_t OFF_SH    = 0;                         // f32 [128*64*512]
static constexpr size_t OFF_SC    = OFF_SH    + 16777216;      // f32 [128*64*512]
static constexpr size_t OFF_HBUF  = OFF_SC    + 16777216;      // f32 [128*64*512]
static constexpr size_t OFF_CBUF  = OFF_HBUF  + 16777216;      // f32 [128*64*512]
static constexpr size_t OFF_SENT  = OFF_CBUF  + 16777216;      // bf16 [128*64*512]
static constexpr size_t OFF_WWORD = OFF_SENT  + 8388608;       // bf16 packed [512*1024]
static constexpr size_t OFF_WLR   = OFF_WWORD + 1048576;       // bf16 packed [1024*2560]
static constexpr size_t OFF_A     = OFF_WLR   + 5242880;       // bf16 [128*1024]
static constexpr size_t OFF_SP    = OFF_A     + 262144;        // int  [2*128]
static constexpr size_t OFF_BP    = OFF_SP    + 1024;          // int  [2*128]
static constexpr size_t OFF_CTR   = OFF_BP    + 1024;          // u32  [512]
static constexpr size_t WS_NEEDED = OFF_CTR   + 2048;

#define SCAN_BLOCKS 32

__device__ inline int clamp63(int x) { x = x < 0 ? 0 : x; return x > 63 ? 63 : x; }
__device__ inline float sigmf(float x) { return 1.0f / (1.0f + __expf(-x)); }
__device__ inline v16bf cat16(v8bf lo, v8bf hi) {
  return __builtin_shufflevector(lo, hi, 0,1,2,3,4,5,6,7,8,9,10,11,12,13,14,15);
}

// ---------------------------------------------------------------------------
// Init: zero sh/sc, convert sentence->bf16, pack W_word and [W_left;W_right]
// into the WMMA B-fragment lane layout:
//   pack[((nn*KT + kk)*32 + lane)*16 + j] = W[k = 32*kk + 16*(j/8) + 8*(lane/16) + j%8][n = 16*nn + lane%16]
// ---------------------------------------------------------------------------
__global__ void spinn_init(const float* __restrict__ sent,
                           const float* __restrict__ Wword,
                           const float* __restrict__ Wl,
                           const float* __restrict__ Wr,
                           float* __restrict__ zero_base,   // sh..sc contiguous (8388608 f32)
                           __bf16* __restrict__ sentb,
                           __bf16* __restrict__ wwp,
                           __bf16* __restrict__ wlrp,
                           int* __restrict__ spb, int* __restrict__ bpb,
                           unsigned* __restrict__ ctr) {
  size_t stride = (size_t)gridDim.x * blockDim.x;
  for (size_t i = (size_t)blockIdx.x * blockDim.x + threadIdx.x; i < 8388608u; i += stride) {
    zero_base[i] = 0.0f;
    if (i < 4194304u) sentb[i] = (__bf16)sent[i];
    if (i < 524288u) {           // W_word pack: KT=16 tiles, NT=64 tiles
      int j = (int)(i & 15), lane = (int)((i >> 4) & 31);
      int rem = (int)(i >> 9);
      int kk = rem & 15, nn = rem >> 4;
      int k = kk * 32 + ((j >> 3) << 4) + ((lane >> 4) << 3) + (j & 7);
      int n = nn * 16 + (lane & 15);
      wwp[i] = (__bf16)Wword[(size_t)k * 1024 + n];
    }
    if (i < 2621440u) {          // [W_left;W_right] pack: KT=32 tiles, NT=160 tiles
      int j = (int)(i & 15), lane = (int)((i >> 4) & 31);
      int rem = (int)(i >> 9);
      int kk = rem & 31, nn = rem >> 5;
      int k = kk * 32 + ((j >> 3) << 4) + ((lane >> 4) << 3) + (j & 7);
      int n = nn * 16 + (lane & 15);
      float v = (k < 512) ? Wl[(size_t)k * 2560 + n] : Wr[(size_t)(k - 512) * 2560 + n];
      wlrp[i] = (__bf16)v;
    }
    if (i < 512u) ctr[i] = 0u;
    if (i < 256u) { spb[i] = 0; bpb[i] = 0; }
  }
}

// ---------------------------------------------------------------------------
// Word projection: (8192 x 512) @ (512 x 1024) -> h_buf | c_buf  (WMMA bf16)
// One 16x16 tile per wave; 32768 tiles = 4096 blocks * 8 waves.
// ---------------------------------------------------------------------------
__global__ void spinn_wordproj(const __bf16* __restrict__ sentb,
                               const __bf16* __restrict__ wwp,
                               const float* __restrict__ bword,
                               float* __restrict__ hbuf,
                               float* __restrict__ cbuf) {
  int tid  = blockIdx.x * blockDim.x + threadIdx.x;
  int wave = tid >> 5, lane = tid & 31;
  int mt = wave & 511, nt = wave >> 9;     // 512 M-tiles, 64 N-tiles
  v8f acc = {};
  const __bf16* arow = sentb + (size_t)(mt * 16 + (lane & 15)) * 512;
  int koff = (lane >> 4) * 8;
#pragma unroll 4
  for (int kk = 0; kk < 16; ++kk) {
    if (kk < 15) __builtin_prefetch((const void*)(arow + (kk + 1) * 32 + koff), 0, 3);
    v8bf lo = *(const v8bf*)(arow + kk * 32 + koff);
    v8bf hi = *(const v8bf*)(arow + kk * 32 + koff + 16);
    v16bf a = cat16(lo, hi);
    v16bf b = *(const v16bf*)(wwp + (((size_t)(nt * 16 + kk) * 32 + lane) << 4));
    acc = __builtin_amdgcn_wmma_f32_16x16x32_bf16(false, a, false, b, (short)0, acc, false, false);
  }
  int n = nt * 16 + (lane & 15);
  float bias = bword[n];
  int rbase = mt * 16 + (lane >> 4) * 8;
#pragma unroll
  for (int r = 0; r < 8; ++r) {
    int m = rbase + r;
    float v = acc[r] + bias;
    if (n < 512) hbuf[(size_t)m * 512 + n]       = v;
    else         cbuf[(size_t)m * 512 + (n-512)] = v;
  }
}

// ---------------------------------------------------------------------------
// Device-wide barrier: fresh counter per use (no reset hazard), sleep-spin.
// ---------------------------------------------------------------------------
__device__ inline void grid_barrier(unsigned* ctr, int idx, unsigned nb) {
  __syncthreads();
  if (threadIdx.x == 0) {
    __threadfence();
    __hip_atomic_fetch_add(&ctr[idx], 1u, __ATOMIC_ACQ_REL, __HIP_MEMORY_SCOPE_AGENT);
    while (__hip_atomic_load(&ctr[idx], __ATOMIC_ACQUIRE, __HIP_MEMORY_SCOPE_AGENT) < nb)
      __builtin_amdgcn_s_sleep(1);
    __threadfence();
  }
  __syncthreads();
}

// ---------------------------------------------------------------------------
// Persistent scan kernel: 127 steps, 2 grid barriers per step.
// Block b owns H-tile ht=b; its 8 waves are the 8 M-tiles. The block's entire
// recurrent-weight working set (5 gates x 32 k-tiles x 1KB = 160KB) is staged
// ONCE into LDS (320KB/WGP -> 2 blocks/WGP) and feeds WMMA from ds_load for
// all 127 steps, eliminating ~5GB of repeated L2 weight traffic.
// ---------------------------------------------------------------------------
__global__ void __launch_bounds__(256)
spinn_scan(const int* __restrict__ trans,
           const float* __restrict__ hbuf, const float* __restrict__ cbuf,
           const __bf16* __restrict__ wlr, const float* __restrict__ bred,
           float* __restrict__ sh, float* __restrict__ sc,
           __bf16* __restrict__ A,
           int* __restrict__ spb, int* __restrict__ bpb,
           unsigned* __restrict__ ctr,
           float* __restrict__ out) {
  __shared__ __align__(32) __bf16 Bs[5 * 32 * 32 * 16];     // 160 KB
  const int tid  = blockIdx.x * blockDim.x + threadIdx.x;   // 0..8191
  const int nth  = SCAN_BLOCKS * 256;
  const int lane = threadIdx.x & 31;
  const int mt   = threadIdx.x >> 5;     // 0..7  (batch tile = local wave)
  const int ht   = blockIdx.x;           // 0..31 (H column tile = block)
  const int hcol  = ht * 16 + (lane & 15);
  const int koff  = (lane >> 4) * 8;
  const int rbase = mt * 16 + (lane >> 4) * 8;
  const __bf16* arow = A + (size_t)(mt * 16 + (lane & 15)) * 1024;

  // ---- One-time LDS stage of this block's 5 gate weight panels ----
  // Pack layout makes each (g, ht) panel a contiguous 32KB chunk.
#pragma unroll
  for (int g = 0; g < 5; ++g) {
    const v8bf* s8 = (const v8bf*)(wlr + (size_t)(g * 32 + ht) * 32 * 512);
    v8bf* d8 = (v8bf*)(Bs + g * 16384);
    for (int i = threadIdx.x; i < 2048; i += 256) d8[i] = s8[i];
  }
  __syncthreads();

  for (int t = 0; t < TN; ++t) {
    const int* sp = spb + (t & 1) * 128;
    const int* bp = bpb + (t & 1) * 128;
    int* spn = spb + ((t + 1) & 1) * 128;
    int* bpn = bpb + ((t + 1) & 1) * 128;
    const int ops_left = TN - t;

    // ---- Phase 1: stage A = [hl, hr] in bf16 (128 x 1024) ----
    for (int e = tid; e < BN * 1024; e += nth) {
      int m = e >> 10, k = e & 1023;
      int s  = sp[m];
      int li = clamp63(s - 2), ri = clamp63(s - 1);
      float v = (k < 512) ? sh[(size_t)(m * 64 + li) * 512 + k]
                          : sh[(size_t)(m * 64 + ri) * 512 + (k - 512)];
      A[e] = (__bf16)v;
    }
    grid_barrier(ctr, 2 * t, SCAN_BLOCKS);

    // ---- Phase 2: gates GEMM (WMMA, B from LDS) + fused TreeLSTM ----
    v8f acc0 = {}, acc1 = {}, acc2 = {}, acc3 = {}, acc4 = {};
    for (int kk = 0; kk < 32; ++kk) {
      if (kk < 31) __builtin_prefetch((const void*)(arow + (kk + 1) * 32 + koff), 0, 3);
      v8bf lo = *(const v8bf*)(arow + kk * 32 + koff);
      v8bf hi = *(const v8bf*)(arow + kk * 32 + koff + 16);
      v16bf a = cat16(lo, hi);
      const int bo = (kk * 32 + lane) * 16;
      v16bf b0 = *(const v16bf*)(Bs + 0 * 16384 + bo);
      v16bf b1 = *(const v16bf*)(Bs + 1 * 16384 + bo);
      v16bf b2 = *(const v16bf*)(Bs + 2 * 16384 + bo);
      v16bf b3 = *(const v16bf*)(Bs + 3 * 16384 + bo);
      v16bf b4 = *(const v16bf*)(Bs + 4 * 16384 + bo);
      acc0 = __builtin_amdgcn_wmma_f32_16x16x32_bf16(false, a, false, b0, (short)0, acc0, false, false);
      acc1 = __builtin_amdgcn_wmma_f32_16x16x32_bf16(false, a, false, b1, (short)0, acc1, false, false);
      acc2 = __builtin_amdgcn_wmma_f32_16x16x32_bf16(false, a, false, b2, (short)0, acc2, false, false);
      acc3 = __builtin_amdgcn_wmma_f32_16x16x32_bf16(false, a, false, b3, (short)0, acc3, false, false);
      acc4 = __builtin_amdgcn_wmma_f32_16x16x32_bf16(false, a, false, b4, (short)0, acc4, false, false);
    }
    const float br0 = bred[0 * 512 + hcol];
    const float br1 = bred[1 * 512 + hcol];
    const float br2 = bred[2 * 512 + hcol];
    const float br3 = bred[3 * 512 + hcol];
    const float br4 = bred[4 * 512 + hcol];

#pragma unroll
    for (int r = 0; r < 8; ++r) {
      int m = rbase + r;
      int s = sp[m];
      int a_ = trans[m * TN + t];
      if (s > ops_left) a_ = 2;                // forced REDUCE
      if (a_ == 2) {                           // REDUCE
        int li = clamp63(s - 2), ri = clamp63(s - 1);
        float cl = sc[(size_t)(m * 64 + li) * 512 + hcol];
        float cr = sc[(size_t)(m * 64 + ri) * 512 + hcol];
        float gg = acc0[r] + br0, ii = acc1[r] + br1;
        float f1 = acc2[r] + br2, f2 = acc3[r] + br3, oo = acc4[r] + br4;
        float c_red = tanhf(gg) * sigmf(ii) + sigmf(f1) * cl + sigmf(f2) * cr;
        float h_red = sigmf(oo) * tanhf(c_red);
        sh[(size_t)(m * 64 + li) * 512 + hcol] = h_red;
        sc[(size_t)(m * 64 + li) * 512 + hcol] = c_red;
      } else {                                 // SHIFT: push from buffer
        int wi = clamp63(s), bi = clamp63(bp[m]);
        sh[(size_t)(m * 64 + wi) * 512 + hcol] = hbuf[(size_t)(m * 64 + bi) * 512 + hcol];
        sc[(size_t)(m * 64 + wi) * 512 + hcol] = cbuf[(size_t)(m * 64 + bi) * 512 + hcol];
      }
    }

    // pointer update (single writer: block 0, threads 0..127)
    if (blockIdx.x == 0 && threadIdx.x < 128) {
      int m = threadIdx.x;
      int s = sp[m];
      int a_ = trans[m * TN + t];
      if (s > ops_left) a_ = 2;
      int shf = (a_ == 1), red = (a_ == 2);
      spn[m] = s + shf - red;
      bpn[m] = bp[m] + shf;
    }
    grid_barrier(ctr, 2 * t + 1, SCAN_BLOCKS);
  }

  // ---- Output gather: out[m][h] = sh[m][clip(sp-1)][h] ----
  const int* spf = spb + (TN & 1) * 128;
  for (int e = tid; e < BN * HN; e += nth) {
    int m = e >> 9, h = e & 511;
    int idx = clamp63(spf[m] - 1);
    out[e] = sh[(size_t)(m * 64 + idx) * 512 + h];
  }
}

// ---------------------------------------------------------------------------
extern "C" void kernel_launch(void* const* d_in, const int* in_sizes, int n_in,
                              void* d_out, int out_size, void* d_ws, size_t ws_size,
                              hipStream_t stream) {
  if (ws_size < WS_NEEDED) return;
  const float* sentence = (const float*)d_in[0];
  const int*   trans    = (const int*)  d_in[1];
  const float* W_word   = (const float*)d_in[2];
  const float* b_word   = (const float*)d_in[3];
  const float* W_left   = (const float*)d_in[4];
  const float* W_right  = (const float*)d_in[5];
  const float* b_reduce = (const float*)d_in[6];
  char* ws = (char*)d_ws;

  float*    sh    = (float*)   (ws + OFF_SH);
  float*    sc    = (float*)   (ws + OFF_SC);
  float*    hbuf  = (float*)   (ws + OFF_HBUF);
  float*    cbuf  = (float*)   (ws + OFF_CBUF);
  __bf16*   sentb = (__bf16*)  (ws + OFF_SENT);
  __bf16*   wwp   = (__bf16*)  (ws + OFF_WWORD);
  __bf16*   wlrp  = (__bf16*)  (ws + OFF_WLR);
  __bf16*   Abuf  = (__bf16*)  (ws + OFF_A);
  int*      spb   = (int*)     (ws + OFF_SP);
  int*      bpb   = (int*)     (ws + OFF_BP);
  unsigned* ctr   = (unsigned*)(ws + OFF_CTR);
  float*    out   = (float*)   d_out;

  spinn_init<<<2048, 256, 0, stream>>>(sentence, W_word, W_left, W_right,
                                       sh /* sh..sc contiguous */, sentb, wwp, wlrp,
                                       spb, bpb, ctr);
  spinn_wordproj<<<4096, 256, 0, stream>>>(sentb, wwp, b_word, hbuf, cbuf);
  spinn_scan<<<SCAN_BLOCKS, 256, 0, stream>>>(trans, hbuf, cbuf, wlrp, b_reduce,
                                              sh, sc, Abuf, spb, bpb, ctr, out);
}
